// Q_Mlp_360777253089
// MI455X (gfx1250) — 8-bit matrix operand order repaired and checked by reading, compiled, not run on hardware
//
#include <hip/hip_runtime.h>
#include <hip/hip_bf16.h>
#include <math.h>

#define M_TOK 16384
#define DDIM  768
#define HDIM  3072

typedef __attribute__((ext_vector_type(8))) int v8i;

__device__ __forceinline__ v8i pack8(int a,int b,int c,int d,int e,int f,int g,int h){
  v8i v; v[0]=a; v[1]=b; v[2]=c; v[3]=d; v[4]=e; v[5]=f; v[6]=g; v[7]=h; return v;
}

__global__ void qmlp_means_kernel(const float* __restrict__ a1, int n1,
                                  const float* __restrict__ a2, int n2,
                                  float* __restrict__ sc) {
  __shared__ float red[256];
  const int t = threadIdx.x;
  float s1 = 0.f, s2 = 0.f;
  for (int i = t; i < n1; i += 256) s1 += a1[i];
  for (int i = t; i < n2; i += 256) s2 += a2[i];
  red[t] = s1; __syncthreads();
  for (int off = 128; off > 0; off >>= 1) { if (t < off) red[t] += red[t + off]; __syncthreads(); }
  if (t == 0) sc[0] = red[0] / (float)n1;
  __syncthreads();
  red[t] = s2; __syncthreads();
  for (int off = 128; off > 0; off >>= 1) { if (t < off) red[t] += red[t + off]; __syncthreads(); }
  if (t == 0) sc[1] = red[0] / (float)n2;
}

__device__ __forceinline__ signed char q4sat(float x, float inv) {
  float tq = fminf(fmaxf(x * inv, -8.0f), 7.0f);
  return (signed char)(int)rintf(tq);
}

__global__ void qmlp_quant_act_kernel(const float* __restrict__ x,
                                      const float* __restrict__ sc,
                                      signed char* __restrict__ q, int n4) {
  int i = blockIdx.x * blockDim.x + threadIdx.x;
  if (i >= n4) return;
  const float inv = 1.0f / sc[0];
  float4 v = ((const float4*)x)[i];
  signed char* o = q + (size_t)i * 4;
  o[0] = q4sat(v.x, inv); o[1] = q4sat(v.y, inv);
  o[2] = q4sat(v.z, inv); o[3] = q4sat(v.w, inv);
}

__global__ void qmlp_quant_w_kernel(const float* __restrict__ w,
                                    const float* __restrict__ alpha,
                                    signed char* __restrict__ q,
                                    int rows, int cols) {
  int i = blockIdx.x * blockDim.x + threadIdx.x;
  int n4 = (rows * cols) >> 2;
  if (i >= n4) return;
  int r = (i * 4) / cols;
  const float inv = 1.0f / alpha[r];
  float4 v = ((const float4*)w)[i];
  signed char* o = q + (size_t)i * 4;
  o[0] = q4sat(v.x, inv); o[1] = q4sat(v.y, inv);
  o[2] = q4sat(v.z, inv); o[3] = q4sat(v.w, inv);
}

template <int EPI>
__global__ __launch_bounds__(256)
void qmlp_gemm_wmma_kernel(const signed char* __restrict__ A,
                           const signed char* __restrict__ W,
                           void* __restrict__ Out,
                           const float* __restrict__ alpha,
                           const float* __restrict__ bias,
                           const float* __restrict__ sc,
                           int M, int N, int K) {
  const int lane  = threadIdx.x & 31;
  const int wave  = threadIdx.x >> 5;
  const int waveM = wave >> 1;
  const int waveN = wave & 1;
  const int mBase = blockIdx.y * 128 + waveM * 32;
  const int nBase = blockIdx.x * 128 + waveN * 64;
  const int l16   = lane & 15;
  const int half  = lane >> 4;

  v8i acc[2][4] = {};

  const signed char* pa0 = A + (size_t)(mBase + l16) * K + half * 16;
  const signed char* pa1 = A + (size_t)(mBase + 16 + l16) * K + half * 16;
  const signed char* pb0 = W + (size_t)(nBase + 0 * 16 + l16) * K + half * 16;
  const signed char* pb1 = W + (size_t)(nBase + 1 * 16 + l16) * K + half * 16;
  const signed char* pb2 = W + (size_t)(nBase + 2 * 16 + l16) * K + half * 16;
  const signed char* pb3 = W + (size_t)(nBase + 3 * 16 + l16) * K + half * 16;
  const signed char* pb[4] = {pb0, pb1, pb2, pb3};

  for (int k0 = 0; k0 < K; k0 += 64) {
    if (k0 + 64 < K) {
      __builtin_prefetch(pa0 + k0 + 64, 0, 3);
      __builtin_prefetch(pa1 + k0 + 64, 0, 3);
      __builtin_prefetch(pb[0] + k0 + 64, 0, 3);
      __builtin_prefetch(pb[2] + k0 + 64, 0, 3);
    }
    v8i af[2], bf[4];
    {
      const signed char* p = pa0 + k0;
      int2 d0 = *(const int2*)(p);
      int2 d1 = *(const int2*)(p + 8);
      int2 d2 = *(const int2*)(p + 32);
      int2 d3 = *(const int2*)(p + 40);
      af[0] = pack8(d0.x, d0.y, d1.x, d1.y, d2.x, d2.y, d3.x, d3.y);
      p = pa1 + k0;
      d0 = *(const int2*)(p);      d1 = *(const int2*)(p + 8);
      d2 = *(const int2*)(p + 32); d3 = *(const int2*)(p + 40);
      af[1] = pack8(d0.x, d0.y, d1.x, d1.y, d2.x, d2.y, d3.x, d3.y);
    }
#pragma unroll
    for (int j = 0; j < 4; ++j) {
      const signed char* p = pb[j] + k0;
      int4 lo = *(const int4*)(p);
      int4 hi = *(const int4*)(p + 32);
      bf[j] = pack8(lo.x, lo.y, lo.z, lo.w, hi.x, hi.y, hi.z, hi.w);
    }
#pragma unroll
    for (int t = 0; t < 2; ++t)
#pragma unroll
      for (int j = 0; j < 4; ++j)
        acc[t][j] = __builtin_amdgcn_wmma_i32_16x16x64_iu8(
             true, af[t],  true, bf[j], acc[t][j],
             false,  false);
  }

  if (EPI == 1) {
    signed char* Hq = (signed char*)Out;
    const float a1m = sc[0];
    const float inv_a2 = 1.0f / sc[1];
#pragma unroll
    for (int j = 0; j < 4; ++j) {
      const int n = nBase + j * 16 + l16;
      const float s = alpha[n] * a1m;
      const float b = bias[n];
#pragma unroll
      for (int t = 0; t < 2; ++t) {
        const int mrow = mBase + t * 16 + half * 8;
#pragma unroll
        for (int r = 0; r < 8; ++r) {
          float v = (float)acc[t][j][r] * s + b;
          float g = 0.5f * v * (1.0f + erff(v * 0.70710678118654752f));
          g = fminf(fmaxf(g, -10.0f), 10.0f);
          float tq = fminf(fmaxf(g * inv_a2, -8.0f), 7.0f);
          Hq[(size_t)(mrow + r) * N + n] = (signed char)(int)rintf(tq);
        }
      }
    }
  } else {
    float* O = (float*)Out;
    const float a2m = sc[1];
#pragma unroll
    for (int j = 0; j < 4; ++j) {
      const int n = nBase + j * 16 + l16;
      const float s = alpha[n] * a2m;
      const float b = bias[n];
#pragma unroll
      for (int t = 0; t < 2; ++t) {
        const int mrow = mBase + t * 16 + half * 8;
#pragma unroll
        for (int r = 0; r < 8; ++r)
          O[(size_t)(mrow + r) * N + n] = (float)acc[t][j][r] * s + b;
      }
    }
  }
}

extern "C" void kernel_launch(void* const* d_in, const int* in_sizes, int n_in,
                              void* d_out, int out_size, void* d_ws, size_t ws_size,
                              hipStream_t stream) {
  const float* x0       = (const float*)d_in[0];
  const float* w1       = (const float*)d_in[1];
  const float* b1       = (const float*)d_in[2];
  const float* alpha_w1 = (const float*)d_in[3];
  const float* alpha_a1 = (const float*)d_in[4];
  const float* w2       = (const float*)d_in[5];
  const float* b2       = (const float*)d_in[6];
  const float* alpha_w2 = (const float*)d_in[7];
  const float* alpha_a2 = (const float*)d_in[8];
  float* out            = (float*)d_out;

  char* ws = (char*)d_ws;
  float*       sc  = (float*)ws;
  signed char* aq  = (signed char*)(ws + 256);
  signed char* wq1 = aq  + (size_t)M_TOK * DDIM;
  signed char* wq2 = wq1 + (size_t)HDIM * DDIM;
  signed char* hq  = wq2 + (size_t)DDIM * HDIM;
  (void)in_sizes; (void)n_in; (void)out_size; (void)ws_size;

  qmlp_means_kernel<<<1, 256, 0, stream>>>(alpha_a1, DDIM, alpha_a2, HDIM, sc);

  {
    int n4 = (M_TOK * DDIM) / 4;
    qmlp_quant_act_kernel<<<(n4 + 255) / 256, 256, 0, stream>>>(x0, sc, aq, n4);
  }
  {
    int n4 = (HDIM * DDIM) / 4;
    qmlp_quant_w_kernel<<<(n4 + 255) / 256, 256, 0, stream>>>(w1, alpha_w1, wq1, HDIM, DDIM);
    qmlp_quant_w_kernel<<<(n4 + 255) / 256, 256, 0, stream>>>(w2, alpha_w2, wq2, DDIM, HDIM);
  }

  {
    dim3 grid(HDIM / 128, M_TOK / 128);
    qmlp_gemm_wmma_kernel<1><<<grid, 256, 0, stream>>>(aq, wq1, hq, alpha_w1, b1, sc,
                                                       M_TOK, HDIM, DDIM);
  }
  {
    dim3 grid(DDIM / 128, M_TOK / 128);
    qmlp_gemm_wmma_kernel<2><<<grid, 256, 0, stream>>>(hq, wq2, out, alpha_w2, b2, sc,
                                                       M_TOK, DDIM, HDIM);
  }
}
